// CausalSelfAttention_90142773609112
// MI455X (gfx1250) — matrix operand order repaired and checked by reading, not compiled, not run on hardware
//
#include <hip/hip_runtime.h>
#include <hip/hip_bf16.h>

// ---------------- problem constants (match reference) ----------------
#define B_        2
#define T_        2048
#define C_        1024
#define NH_FULL   8
#define NH_RED    8
#define HD_       64
#define RD_       32
#define WIN_FULL  1024
#define WIN_RED   256
#define TPAD      (T_ + 32)          // pad V^T rows so 32-wide key tail blocks stay in-bounds
#define MTOT      (B_ * T_)          // 4096 token rows

typedef __attribute__((ext_vector_type(16))) __bf16         v16bf;
typedef __attribute__((ext_vector_type(8)))  float          v8f;
typedef __attribute__((ext_vector_type(16))) unsigned short v16u;
typedef __attribute__((ext_vector_type(8)))  unsigned short v8u;

union FragU { v16u u; v16bf b; };

// fp32 -> bf16 round-to-nearest-even (finite inputs)
__device__ __forceinline__ unsigned short f2bf(float f) {
  unsigned int u = __float_as_uint(f);
  u += 0x7FFFu + ((u >> 16) & 1u);
  return (unsigned short)(u >> 16);
}

// two contiguous 16B bf16 loads -> one 16-element fragment
__device__ __forceinline__ v16bf load_frag(const unsigned short* p0, const unsigned short* p1) {
  v8u lo = *(const v8u*)p0;
  v8u hi = *(const v8u*)p1;
  v16u u = __builtin_shufflevector(lo, hi, 0,1,2,3,4,5,6,7,8,9,10,11,12,13,14,15);
  FragU f; f.u = u; return f.b;
}

__device__ __forceinline__ v8f wmma_bf16(v16bf a, v16bf b, v8f c) {
  return __builtin_amdgcn_wmma_f32_16x16x32_bf16(false, a, false, b, (short)0, c, false, false);
}

// reductions across the 16 lanes of a half-wave (C/D rows live per half)
__device__ __forceinline__ float redmax16(float v) {
  v = fmaxf(v, __shfl_xor(v, 1, 32));
  v = fmaxf(v, __shfl_xor(v, 2, 32));
  v = fmaxf(v, __shfl_xor(v, 4, 32));
  v = fmaxf(v, __shfl_xor(v, 8, 32));
  return v;
}
__device__ __forceinline__ float redsum16(float v) {
  v += __shfl_xor(v, 1, 32);
  v += __shfl_xor(v, 2, 32);
  v += __shfl_xor(v, 4, 32);
  v += __shfl_xor(v, 8, 32);
  return v;
}

// ---------------- conversion kernels ----------------
__global__ void k_f32_to_bf16(const float* __restrict__ in, unsigned short* __restrict__ out, int n) {
  int i = blockIdx.x * blockDim.x + threadIdx.x;
  if (i < n) out[i] = f2bf(in[i]);
}

// in: fp32 [K][N] row-major  ->  out: bf16 [N][K] (transposed)
__global__ void k_transpose_bf16(const float* __restrict__ in, unsigned short* __restrict__ out,
                                 int K, int N) {
  int i = blockIdx.x * blockDim.x + threadIdx.x;
  if (i < K * N) {
    int k = i / N, n = i % N;
    out[(size_t)n * K + k] = f2bf(in[i]);
  }
}

// ---------------- bf16 WMMA GEMM with async-copy B staging ----------------
// Block = 128 threads = 4 waves stacked in M; all share one 64-col B slab.
// Per K-step (32): each thread async-copies the 32 contiguous bytes of one
// B-fragment lane into fragment-ordered LDS (double-buffered, ASYNCcnt),
// overlapping the copy of step k+1 with the 8 WMMAs of step k.
// A: bf16 [M][K] row-major.  Wt: bf16 [N][K] (weights pre-transposed).
// mode 0: bf16 row-major out (ldo).  mode 1: bf16 V^T per head: out[(b*hd+n)*TPAD + t].
// mode 2: fp32 row-major out (ldo).
__global__ void __launch_bounds__(128)
k_gemm_bf16(const unsigned short* __restrict__ A, int lda,
            const unsigned short* __restrict__ Wt, int ldw,
            int K, int mode, void* __restrict__ outp, int ldo, int hd) {
  __shared__ unsigned short sB[2][4][32][16];   // [buf][frag j][lane][16 elems] = 8KB

  const int tid  = threadIdx.x;
  const int wave = tid >> 5;
  const int lane = tid & 31;
  const int lh = lane >> 4;      // half-wave id
  const int ln = lane & 15;
  const int mBase = blockIdx.y * 128 + wave * 32;
  const int nBase = blockIdx.x * 64;

  // This thread stages B-fragment j=wave, fragment-lane=lane:
  // 16 contiguous bf16 at Wt[(nBase+16j+ln)*ldw + k0 + 16*lh]  (32 bytes)
  const unsigned short* wSrc = Wt + (size_t)(nBase + 16 * wave + ln) * ldw + 16 * lh;
  const unsigned int ldsDst = (unsigned int)(size_t)&sB[0][wave][lane][0];
  const unsigned int bufStride = (unsigned int)sizeof(sB) / 2;   // 4096 bytes

  v8f acc[2][4];
#pragma unroll
  for (int i = 0; i < 2; ++i)
#pragma unroll
    for (int j = 0; j < 4; ++j) acc[i][j] = (v8f){};

  const unsigned short* aRow0 = A + (size_t)(mBase + ln) * lda;
  const unsigned short* aRow1 = A + (size_t)(mBase + 16 + ln) * lda;

  // prologue: stage buffer 0
  {
    const unsigned short* g = wSrc;
    unsigned int l = ldsDst;
    asm volatile(
      "global_load_async_to_lds_b128 %0, %1, off\n\t"
      "global_load_async_to_lds_b128 %0, %1, off offset:16"
      :: "v"(l), "v"(g) : "memory");
  }
  asm volatile("s_wait_asynccnt 0x0" ::: "memory");
  __syncthreads();

  for (int k0 = 0; k0 < K; k0 += 32) {
    const int buf = (k0 >> 5) & 1;
    const bool more = (k0 + 32 < K);
    if (more) {
      // stage next K-step into the other buffer (overlaps with WMMAs below)
      const unsigned short* g = wSrc + k0 + 32;
      unsigned int l = ldsDst + (buf ^ 1) * bufStride;
      asm volatile(
        "global_load_async_to_lds_b128 %0, %1, off\n\t"
        "global_load_async_to_lds_b128 %0, %1, off offset:16"
        :: "v"(l), "v"(g) : "memory");
      __builtin_prefetch(aRow0 + k0 + 32, 0, 1);
      __builtin_prefetch(aRow1 + k0 + 32, 0, 1);
    }
    // A fragments (direct): lane holds row M=ln, K = k0 + 16*lh + e
    v16bf af[2];
    {
      const unsigned short* p = aRow0 + k0 + 16 * lh;
      af[0] = load_frag(p, p + 8);
      const unsigned short* q = aRow1 + k0 + 16 * lh;
      af[1] = load_frag(q, q + 8);
    }
    // B fragments from LDS (fragment-ordered): ds_load_b128 pairs
    v16bf bf_[4];
#pragma unroll
    for (int j = 0; j < 4; ++j) {
      const unsigned short* bp = &sB[buf][j][lane][0];
      bf_[j] = load_frag(bp, bp + 8);
    }
#pragma unroll
    for (int i = 0; i < 2; ++i)
#pragma unroll
      for (int j = 0; j < 4; ++j) acc[i][j] = wmma_bf16(af[i], bf_[j], acc[i][j]);

    if (more) {
      asm volatile("s_wait_asynccnt 0x0" ::: "memory");
      __syncthreads();
    }
  }

  // epilogue: D element r -> (m = r + 8*lh, n = ln)
#pragma unroll
  for (int i = 0; i < 2; ++i)
#pragma unroll
    for (int j = 0; j < 4; ++j)
#pragma unroll
      for (int r = 0; r < 8; ++r) {
        int m = mBase + 16 * i + r + 8 * lh;
        int n = nBase + 16 * j + ln;
        float v = acc[i][j][r];
        if (mode == 0) {
          ((unsigned short*)outp)[(size_t)m * ldo + n] = f2bf(v);
        } else if (mode == 1) {
          int b = m / T_, t = m % T_;
          ((unsigned short*)outp)[((size_t)b * hd + n) * TPAD + t] = f2bf(v);
        } else {
          ((float*)outp)[(size_t)m * ldo + n] = v;
        }
      }
}

// ---------------- windowed causal attention (flash-style), one wave per 16-query tile ----------------
// qk: bf16 [B*T][ldqk] holding q cols at qOff + h*d, k cols at kOff + h*d.
// vt: bf16 [(b*H + h)*64][TPAD]  (V transposed per head, dv = 64)
// y : bf16 [B*T][ldy], head output at yOff + h*64.
__global__ void __launch_bounds__(32)
k_attn_win(const unsigned short* __restrict__ qk, int ldqk, int qOff, int kOff,
           int d, int win, const unsigned short* __restrict__ vt,
           unsigned short* __restrict__ y, int ldy, int yOff, int H, float scl) {
  __shared__ float sP[16][36];   // P tile staging (C-layout -> A-layout transpose)

  const int lane = threadIdx.x & 31;
  const int lh = lane >> 4;
  const int ln = lane & 15;
  const int m0 = blockIdx.x * 16;           // query tile start (t index)
  const int bh = blockIdx.y;                // b*H + h
  const int b = bh / H, h = bh % H;
  const size_t rowBase = (size_t)b * T_ + m0;
  const int nd = d >> 5;                    // 32-wide d chunks (1 or 2)

  // Q fragments (loaded once): lane row = m0+ln, K = dc*32 + 16*lh + e
  v16bf aQ[2];
#pragma unroll
  for (int dc = 0; dc < 2; ++dc) {
    if (dc < nd) {
      const unsigned short* p = qk + (rowBase + ln) * ldqk + qOff + h * d + dc * 32 + 16 * lh;
      aQ[dc] = load_frag(p, p + 8);
    }
  }
  const unsigned short* vbase = vt + (size_t)bh * 64 * TPAD;

  v8f o0 = (v8f){}, o1 = (v8f){}, o2 = (v8f){}, o3 = (v8f){};
  float mrow[8], lsum[8];
#pragma unroll
  for (int r = 0; r < 8; ++r) { mrow[r] = -1e30f; lsum[r] = 0.f; }

  int ks = m0 - win + 1; if (ks < 0) ks = 0; ks &= ~31;

  for (int kb = ks; kb <= m0 + 15; kb += 32) {
    // ---- S = Q K^T for 32 keys (two 16-key B fragments) ----
    v8f S0 = (v8f){}, S1 = (v8f){};
    for (int dc = 0; dc < nd; ++dc) {
      int key0 = kb + ln;      if (key0 > T_ - 1) key0 = T_ - 1;
      int key1 = kb + 16 + ln; if (key1 > T_ - 1) key1 = T_ - 1;
      const unsigned short* kp0 = qk + ((size_t)b * T_ + key0) * ldqk + kOff + h * d + dc * 32 + 16 * lh;
      const unsigned short* kp1 = qk + ((size_t)b * T_ + key1) * ldqk + kOff + h * d + dc * 32 + 16 * lh;
      v16bf bk0 = load_frag(kp0, kp0 + 8);
      v16bf bk1 = load_frag(kp1, kp1 + 8);
      S0 = wmma_bf16(aQ[dc], bk0, S0);
      S1 = wmma_bf16(aQ[dc], bk1, S1);
    }

    // ---- online softmax update (rows r+8*lh, cols kb+ln / kb+16+ln) ----
    float p0v[8], p1v[8];
    const int col0 = kb + ln, col1 = col0 + 16;
#pragma unroll
    for (int r = 0; r < 8; ++r) {
      int i = m0 + r + 8 * lh;
      int d0 = i - col0, d1 = i - col1;
      bool v0 = (d0 >= 0) && (d0 < win);
      bool v1 = (d1 >= 0) && (d1 < win);
      float ms0 = v0 ? S0[r] * scl : -1e30f;
      float ms1 = v1 ? S1[r] * scl : -1e30f;
      float mx = redmax16(fmaxf(ms0, ms1));
      float mnew = fmaxf(mrow[r], mx);
      float scale = __expf(mrow[r] - mnew);
      mrow[r] = mnew;
      float e0 = v0 ? __expf(ms0 - mnew) : 0.f;
      float e1 = v1 ? __expf(ms1 - mnew) : 0.f;
      lsum[r] = lsum[r] * scale + redsum16(e0 + e1);
      p0v[r] = e0; p1v[r] = e1;
      o0[r] *= scale; o1[r] *= scale; o2[r] *= scale; o3[r] *= scale;
    }

    // ---- transpose P from C-layout to A-layout through LDS, cast to bf16 ----
    __syncthreads();
#pragma unroll
    for (int r = 0; r < 8; ++r) {
      sP[r + 8 * lh][ln]      = p0v[r];
      sP[r + 8 * lh][16 + ln] = p1v[r];
    }
    __syncthreads();
    FragU fp;
#pragma unroll
    for (int e = 0; e < 16; ++e) {
      int kk = e + 16 * lh;
      fp.u[e] = f2bf(sP[ln][kk]);
    }
    v16bf aP = fp.b;

    // ---- O += P x V  (V^T rows contiguous along t; padded, no clamp needed) ----
    {
      const unsigned short* v0p = vbase + (size_t)(0 * 16 + ln) * TPAD + kb + 16 * lh;
      o0 = wmma_bf16(aP, load_frag(v0p, v0p + 8), o0);
      const unsigned short* v1p = vbase + (size_t)(1 * 16 + ln) * TPAD + kb + 16 * lh;
      o1 = wmma_bf16(aP, load_frag(v1p, v1p + 8), o1);
      const unsigned short* v2p = vbase + (size_t)(2 * 16 + ln) * TPAD + kb + 16 * lh;
      o2 = wmma_bf16(aP, load_frag(v2p, v2p + 8), o2);
      const unsigned short* v3p = vbase + (size_t)(3 * 16 + ln) * TPAD + kb + 16 * lh;
      o3 = wmma_bf16(aP, load_frag(v3p, v3p + 8), o3);
    }
  }

  // ---- normalize and store y (bf16, row-major) ----
#pragma unroll
  for (int r = 0; r < 8; ++r) {
    float inv = 1.0f / lsum[r];
    size_t row = rowBase + r + 8 * lh;
    unsigned short* yr = y + row * ldy + yOff + h * 64;
    yr[ln]      = f2bf(o0[r] * inv);
    yr[16 + ln] = f2bf(o1[r] * inv);
    yr[32 + ln] = f2bf(o2[r] * inv);
    yr[48 + ln] = f2bf(o3[r] * inv);
  }
}

// ---------------- launcher ----------------
extern "C" void kernel_launch(void* const* d_in, const int* in_sizes, int n_in,
                              void* d_out, int out_size, void* d_ws, size_t ws_size,
                              hipStream_t stream) {
  const float* x      = (const float*)d_in[0];   // [B,T,C]
  const float* w_qkv  = (const float*)d_in[1];   // [C, 1536]
  const float* w_qk   = (const float*)d_in[2];   // [C, 512]
  const float* w_v    = (const float*)d_in[3];   // [C, 512]
  const float* w_proj = (const float*)d_in[4];   // [C, C]
  float* out = (float*)d_out;

  unsigned short* p = (unsigned short*)d_ws;
  unsigned short* xb     = p; p += (size_t)MTOT * C_;        // x bf16
  unsigned short* wqkvT  = p; p += (size_t)1536 * C_;        // W^T
  unsigned short* wqkT   = p; p += (size_t)512 * C_;
  unsigned short* wvT    = p; p += (size_t)512 * C_;
  unsigned short* wprojT = p; p += (size_t)C_ * C_;
  unsigned short* qkf    = p; p += (size_t)MTOT * 1024;      // q(0-511) k(512-1023) full heads
  unsigned short* vtf    = p; p += (size_t)B_ * 512 * TPAD;  // V^T full heads
  unsigned short* qkr    = p; p += (size_t)MTOT * 512;       // q(0-255) k(256-511) reduced heads
  unsigned short* vtr    = p; p += (size_t)B_ * 512 * TPAD;  // V^T reduced heads
  unsigned short* yb     = p; p += (size_t)MTOT * C_;        // attention output bf16

  const int thr = 256;
  k_f32_to_bf16<<<(MTOT * C_ + thr - 1) / thr, thr, 0, stream>>>(x, xb, MTOT * C_);
  k_transpose_bf16<<<(C_ * 1536 + thr - 1) / thr, thr, 0, stream>>>(w_qkv, wqkvT, C_, 1536);
  k_transpose_bf16<<<(C_ * 512 + thr - 1) / thr, thr, 0, stream>>>(w_qk, wqkT, C_, 512);
  k_transpose_bf16<<<(C_ * 512 + thr - 1) / thr, thr, 0, stream>>>(w_v, wvT, C_, 512);
  k_transpose_bf16<<<(C_ * C_ + thr - 1) / thr, thr, 0, stream>>>(w_proj, wprojT, C_, C_);

  // projections (bf16 WMMA GEMMs, async-staged B, 4-wave blocks over 128 M-rows)
  k_gemm_bf16<<<dim3(1024 / 64, MTOT / 128), 128, 0, stream>>>(xb, C_, wqkvT, C_, C_, 0, qkf, 1024, 0);
  k_gemm_bf16<<<dim3(512 / 64, MTOT / 128), 128, 0, stream>>>(xb, C_, wqkvT + (size_t)1024 * C_, C_, C_, 1, vtf, 0, 512);
  k_gemm_bf16<<<dim3(512 / 64, MTOT / 128), 128, 0, stream>>>(xb, C_, wqkT, C_, C_, 0, qkr, 512, 0);
  k_gemm_bf16<<<dim3(512 / 64, MTOT / 128), 128, 0, stream>>>(xb, C_, wvT, C_, C_, 1, vtr, 0, 512);

  // windowed attention: full heads (d=64, win=1024) -> y cols 0-511; reduced (d=32, win=256) -> 512-1023
  k_attn_win<<<dim3(T_ / 16, B_ * NH_FULL), 32, 0, stream>>>(qkf, 1024, 0, 512, 64, WIN_FULL,
                                                             vtf, yb, C_, 0, NH_FULL, 0.125f);
  k_attn_win<<<dim3(T_ / 16, B_ * NH_RED), 32, 0, stream>>>(qkr, 512, 0, 256, 32, WIN_RED,
                                                            vtr, yb, C_, 512, NH_RED, 0.17677669529663687f);

  // output projection -> fp32 d_out
  k_gemm_bf16<<<dim3(C_ / 64, MTOT / 128), 128, 0, stream>>>(yb, C_, wprojT, C_, C_, 2, out, C_, 0);
}
